// ConvLSTM_25091198943444
// MI455X (gfx1250) — matrix operand order repaired and checked by reading, not compiled, not run on hardware
//
#include <hip/hip_runtime.h>

typedef __attribute__((ext_vector_type(16))) _Float16 v16h;
typedef __attribute__((ext_vector_type(8)))  float    v8f;

#define NB    4      // batch
#define NC    16     // input channels
#define NT    32     // time steps
#define NHW   4096   // H*W
#define NU    32     // units
#define NBB   64     // backbone width
#define KIN   48     // C + U
#define ROWP  72     // LDS row pad (halfs) to spread banks

union H2 { unsigned int u; _Float16 h[2]; };

#if __has_builtin(__builtin_amdgcn_tanhf)
// gfx1250 hardware transcendental: v_tanh_f32 (single TRANS op)
__device__ __forceinline__ float fast_tanh(float x) {
    return __builtin_amdgcn_tanhf(x);
}
__device__ __forceinline__ float fast_sigmoid(float x) {
    // sigmoid(x) = 0.5*tanh(x/2) + 0.5  -> mul + v_tanh + fma
    return __builtin_fmaf(0.5f, __builtin_amdgcn_tanhf(0.5f * x), 0.5f);
}
#else
__device__ __forceinline__ float fast_tanh(float x) {
    // tanh(x) = 1 - 2/(exp(2x)+1); saturates correctly at +/-1
    return 1.0f - 2.0f * __builtin_amdgcn_rcpf(1.0f + __expf(2.0f * x));
}
__device__ __forceinline__ float fast_sigmoid(float x) {
    return __builtin_amdgcn_rcpf(1.0f + __expf(-x));
}
#endif

__global__ __launch_bounds__(128)
void cfc_wmma_kernel(const float* __restrict__ x,   const float* __restrict__ Wb,
                     const float* __restrict__ bb,
                     const float* __restrict__ Wff1, const float* __restrict__ bff1,
                     const float* __restrict__ Wff2, const float* __restrict__ bff2,
                     const float* __restrict__ Wta,  const float* __restrict__ bta,
                     const float* __restrict__ Wtb,  const float* __restrict__ btb,
                     float* __restrict__ out)
{
    // head weights (4 heads x 32 rows x 64 cols), rows padded
    __shared__ _Float16 sW[4 * 32 * ROWP];
    // bias fragments in C/D VGPR order: tiles 0..3 = backbone, 4..11 = (head,ut)
    __shared__ float sBias[12 * 16];
    // per-wave state: concat(x_t[0..15], h[0..31], pad[0..15]) as f16
    __shared__ _Float16 sZin[4][16][ROWP];
    // per-wave backbone activations z (64 ch) as f16
    __shared__ _Float16 sZ[4][16][ROWP];

    const int tid  = threadIdx.x;
    const int wave = tid >> 5;
    const int lane = tid & 31;
    const int half = lane >> 4;   // which half-wave (K/M split in WMMA layouts)
    const int col  = lane & 15;   // N index (pixel) / M row within tile

    // ---- one-time staging (workgroup-cooperative) ----
    for (int i = tid; i < 32 * 64; i += 128) {
        int r = i >> 6, k = i & 63;
        sW[(0 * 32 + r) * ROWP + k] = (_Float16)Wff1[i];
        sW[(1 * 32 + r) * ROWP + k] = (_Float16)Wff2[i];
        sW[(2 * 32 + r) * ROWP + k] = (_Float16)Wta[i];
        sW[(3 * 32 + r) * ROWP + k] = (_Float16)Wtb[i];
    }
    for (int i = tid; i < 12 * 16; i += 128) {
        int tile = i >> 4, j = i & 15, hf = j >> 3, r = j & 7;
        float v;
        if (tile < 4) {
            v = bb[tile * 16 + hf * 8 + r];
        } else {
            int hd = (tile - 4) >> 1, ut = (tile - 4) & 1;
            int u = ut * 16 + hf * 8 + r;
            v = (hd == 0) ? bff1[u] : (hd == 1) ? bff2[u] : (hd == 2) ? bta[u] : btb[u];
        }
        sBias[i] = v;
    }
    // zero h-region and pad-region of this wave's input buffer (h0 = 0)
    for (int i = 0; i < 24; ++i)
        sZin[wave][col][16 + half * 24 + i] = (_Float16)0.0f;

    // ---- backbone A fragments in registers (16-bit A-matrix 16x32 layout) ----
    // lane m = col; k(v,e) = kt*32 + (v/4)*16 + half*8 + (v%4)*2 + e ; k>=48 -> 0 pad
    v16h Abb[4][2];
#pragma unroll
    for (int mt = 0; mt < 4; ++mt)
#pragma unroll
        for (int kt = 0; kt < 2; ++kt)
#pragma unroll
            for (int v = 0; v < 8; ++v)
#pragma unroll
                for (int e = 0; e < 2; ++e) {
                    int k = kt * 32 + (v >> 2) * 16 + half * 8 + (v & 3) * 2 + e;
                    float w = (k < KIN) ? Wb[(mt * 16 + col) * KIN + k] : 0.0f;
                    Abb[mt][kt][v * 2 + e] = (_Float16)w;
                }

    __syncthreads();

    const int pix = blockIdx.x * 64 + wave * 16 + col;  // 16 consecutive pixels per wave
    const int b   = pix >> 12;
    const int hw  = pix & (NHW - 1);

    for (int t = 0; t < NT; ++t) {
        // ---- stage x_t into LDS (coalesced: 16 lanes x contiguous hw) ----
#pragma unroll
        for (int i = 0; i < 8; ++i) {
            int c = i * 2 + half;
            const float* gp = &x[((size_t)(b * NC + c) * NT + t) * NHW + hw];
            sZin[wave][col][c] = (_Float16)(*gp);
            if (t + 1 < NT) __builtin_prefetch(gp + NHW, 0, 0);  // next timestep line
        }
        asm volatile("" ::: "memory");  // keep LDS stores before gathers

        // ---- gather input B fragments (K x N layout: lane=pixel, k=(v/4)*16+half*8+(v%4)*2+e) ----
        v16h Bin[2];
#pragma unroll
        for (int kt = 0; kt < 2; ++kt)
#pragma unroll
            for (int v = 0; v < 8; ++v) {
                H2 p;
                p.u = *reinterpret_cast<const unsigned int*>(
                    &sZin[wave][col][kt * 32 + (v >> 2) * 16 + half * 8 + (v & 3) * 2]);
                Bin[kt][v * 2]     = p.h[0];
                Bin[kt][v * 2 + 1] = p.h[1];
            }

        // ---- backbone: z = lecun_tanh(Wb * [x;h] + bb), 4 M-tiles x 2 K-steps ----
#pragma unroll
        for (int mt = 0; mt < 4; ++mt) {
            v8f c;
#pragma unroll
            for (int r = 0; r < 8; ++r) c[r] = sBias[mt * 16 + half * 8 + r];
            v8f acc = __builtin_amdgcn_wmma_f32_16x16x32_f16(
                false, Abb[mt][0], false, Bin[0], (short)0, c, false, false);
            acc = __builtin_amdgcn_wmma_f32_16x16x32_f16(
                false, Abb[mt][1], false, Bin[1], (short)0, acc, false, false);
#pragma unroll
            for (int r = 0; r < 8; ++r)
                sZ[wave][col][mt * 16 + half * 8 + r] =
                    (_Float16)(1.7159f * fast_tanh(0.666f * acc[r]));
        }
        asm volatile("" ::: "memory");

        // ---- gather z B fragments ----
        v16h Bz[2];
#pragma unroll
        for (int kt = 0; kt < 2; ++kt)
#pragma unroll
            for (int v = 0; v < 8; ++v) {
                H2 p;
                p.u = *reinterpret_cast<const unsigned int*>(
                    &sZ[wave][col][kt * 32 + (v >> 2) * 16 + half * 8 + (v & 3) * 2]);
                Bz[kt][v * 2]     = p.h[0];
                Bz[kt][v * 2 + 1] = p.h[1];
            }

        // ---- heads: 2 unit-tiles x 4 heads, A fragments streamed from LDS ----
#pragma unroll
        for (int ut = 0; ut < 2; ++ut) {
            v8f ha[4];
#pragma unroll
            for (int hd = 0; hd < 4; ++hd) {
                v16h A0, A1;
#pragma unroll
                for (int kt = 0; kt < 2; ++kt)
#pragma unroll
                    for (int v = 0; v < 8; ++v) {
                        H2 p;
                        p.u = *reinterpret_cast<const unsigned int*>(
                            &sW[(hd * 32 + ut * 16 + col) * ROWP + kt * 32 +
                                (v >> 2) * 16 + half * 8 + (v & 3) * 2]);
                        if (kt == 0) { A0[v * 2] = p.h[0]; A0[v * 2 + 1] = p.h[1]; }
                        else         { A1[v * 2] = p.h[0]; A1[v * 2 + 1] = p.h[1]; }
                    }
                v8f c;
#pragma unroll
                for (int r = 0; r < 8; ++r)
                    c[r] = sBias[(4 + hd * 2 + ut) * 16 + half * 8 + r];
                v8f a = __builtin_amdgcn_wmma_f32_16x16x32_f16(
                    false, A0, false, Bz[0], (short)0, c, false, false);
                a = __builtin_amdgcn_wmma_f32_16x16x32_f16(
                    false, A1, false, Bz[1], (short)0, a, false, false);
                ha[hd] = a;
            }
            // elementwise combine: h' = tanh(ff1)*(1-s) + s*tanh(ff2), s = sigmoid(ta+tb)
#pragma unroll
            for (int r = 0; r < 8; ++r) {
                float f1 = fast_tanh(ha[0][r]);
                float f2 = fast_tanh(ha[1][r]);
                float ti = fast_sigmoid(ha[2][r] + ha[3][r]);  // ts == 1.0
                float hn = f1 + ti * (f2 - f1);
                int u = ut * 16 + half * 8 + r;
                sZin[wave][col][NC + u] = (_Float16)hn;              // recurrent state
                out[((size_t)(b * NU + u) * NT + t) * NHW + hw] = hn; // (B,U,T,H,W)
            }
        }
        asm volatile("" ::: "memory");
    }
}

extern "C" void kernel_launch(void* const* d_in, const int* in_sizes, int n_in,
                              void* d_out, int out_size, void* d_ws, size_t ws_size,
                              hipStream_t stream) {
    const float* x    = (const float*)d_in[0];
    const float* Wb   = (const float*)d_in[1];
    const float* bb   = (const float*)d_in[2];
    const float* Wff1 = (const float*)d_in[3];
    const float* bff1 = (const float*)d_in[4];
    const float* Wff2 = (const float*)d_in[5];
    const float* bff2 = (const float*)d_in[6];
    const float* Wta  = (const float*)d_in[7];
    const float* bta  = (const float*)d_in[8];
    const float* Wtb  = (const float*)d_in[9];
    const float* btb  = (const float*)d_in[10];
    float* out = (float*)d_out;

    // 16384 pixels total, 64 per workgroup (4 waves x 16 pixels)
    cfc_wmma_kernel<<<256, 128, 0, stream>>>(x, Wb, bb, Wff1, bff1, Wff2, bff2,
                                             Wta, bta, Wtb, btb, out);
}
